// DeltaNet_31877247271504
// MI455X (gfx1250) — compile-verified, matrix operand order checked by reading, not run on hardware
//
#include <hip/hip_runtime.h>
#include <hip/hip_bf16.h>
#include <math.h>

// ---------------- problem constants ----------------
#define BB   4
#define LL   4096
#define DD   1024
#define HH   8
#define DKV  128
#define BL   (BB * LL)      // 16384
#define NCHUNK (LL / 32)    // 128

typedef __attribute__((ext_vector_type(16))) __bf16 v16bf;
typedef __attribute__((ext_vector_type(8)))  float  v8f;
typedef __attribute__((ext_vector_type(4)))  unsigned int u32x4;
typedef __attribute__((ext_vector_type(8)))  int  i32x8;
typedef __attribute__((ext_vector_type(4)))  int  i32x4;

// ---------------------------------------------------------------------------
// Tensor Data Mover: issue a 2D tile load Global -> LDS.
//   tile0 = contiguous elements per row (f32), tile1 = rows,
//   stride0 = row stride in elements. Descriptor per CDNA5 ISA 08_async_tensor.
// Must be called wave-uniform (TDM ignores EXEC; operands live in SGPRs).
// This toolchain exposes the 6-arg builtin: (g0, g1, g2, g3, g4, cpol).
// ---------------------------------------------------------------------------
__device__ __forceinline__ void tdm_load_2d(const float* gptr, unsigned lds_off,
                                            unsigned tile0, unsigned tile1,
                                            unsigned stride0) {
    unsigned long long ga = (unsigned long long)(uintptr_t)gptr;
    u32x4 g0;
    g0[0] = 1u;                                       // count=1, user mode
    g0[1] = lds_off;                                  // lds_addr (bytes)
    g0[2] = (unsigned)(ga & 0xFFFFFFFFu);             // global_addr[31:0]
    g0[3] = (unsigned)((ga >> 32) & 0x01FFFFFFu)      // global_addr[56:32]
          | (2u << 30);                               // type = 2 (image)
    i32x8 g1;
    g1[0] = (int)(2u << 16);                          // data_size = 4 bytes
    g1[1] = (int)((tile0 & 0xFFFFu) << 16);           // tensor_dim0 lo16 -> [31:16]
    g1[2] = (int)((tile0 >> 16) | ((tile1 & 0xFFFFu) << 16)); // dim0 hi / dim1 lo
    g1[3] = (int)((tile1 >> 16) | (tile0 << 16));     // dim1 hi / tile_dim0
    g1[4] = (int)(tile1 & 0xFFFFu);                   // tile_dim1 (tile_dim2 = 0)
    g1[5] = (int)stride0;                             // tensor_dim0_stride lo32
    g1[6] = 0;                                        // stride0 hi / stride1 lo
    g1[7] = 0;
    i32x4 z4 = {0, 0, 0, 0};                          // VADDR2/3: <=2D tensor
    i32x8 z8 = {0, 0, 0, 0, 0, 0, 0, 0};
    __builtin_amdgcn_tensor_load_to_lds(g0, g1, z4, z4, z8, 0);
}
__device__ __forceinline__ unsigned lds_off_of(const void* p) {
    return (unsigned)(uintptr_t)p;                    // addrspace(3) offset
}

// ---------------------------------------------------------------------------
// WMMA fragment loaders from f32 LDS buffers (convert to bf16 in registers).
// A-layout (16x32 bf16): lane<16 -> row M=lane,    K: vgpr p<4 -> 2p,   p>=4 -> 16+2(p-4)
//                        lane>=16 -> row M=lane-16, K: vgpr p<4 -> 8+2p, p>=4 -> 24+2(p-4)
// B-layout is identical with "row" meaning column N.
// ---------------------------------------------------------------------------
__device__ __forceinline__ v16bf frag_rowK(const float* buf, int ld, int r0, int k0, int lane) {
    int row = r0 + (lane & 15);
    int kb  = (lane < 16) ? 0 : 8;
    v16bf f;
#pragma unroll
    for (int p = 0; p < 8; ++p) {
        int kk = (p < 4) ? (kb + 2 * p) : (kb + 8 + 2 * p);
        f[2 * p]     = (__bf16)buf[row * ld + k0 + kk];
        f[2 * p + 1] = (__bf16)buf[row * ld + k0 + kk + 1];
    }
    return f;
}
// elem = buf[(k0+kk)*ld + c0 + (lane&15)]  : row-major [K x N] operand (or transposed-A)
__device__ __forceinline__ v16bf frag_colK(const float* buf, int ld, int k0, int c0, int lane) {
    int col = c0 + (lane & 15);
    int kb  = (lane < 16) ? 0 : 8;
    v16bf f;
#pragma unroll
    for (int p = 0; p < 8; ++p) {
        int kk = (p < 4) ? (kb + 2 * p) : (kb + 8 + 2 * p);
        f[2 * p]     = (__bf16)buf[(k0 + kk) * ld + col];
        f[2 * p + 1] = (__bf16)buf[(k0 + kk + 1) * ld + col];
    }
    return f;
}
__device__ __forceinline__ v8f vzero8() {
    v8f z;
#pragma unroll
    for (int e = 0; e < 8; ++e) z[e] = 0.0f;
    return z;
}

// ---------------------------------------------------------------------------
// Generic tiled GEMM  Y[MxN] = epi(X[MxK] @ W[KxN] + bias)  via bf16 WMMA.
// 128 threads (4 waves), 64x64 tile per WG, K-step 32.
// Tiles staged Global->LDS by the Tensor Data Mover, double buffered:
//   wait tensorcnt(cur) -> barrier -> TDM-prefetch(next) -> compute(cur) -> barrier
// EPI: 0 = none, 1 = exact GELU.
// ---------------------------------------------------------------------------
template <int EPI>
__global__ void __launch_bounds__(128)
gemm_bf16_wmma(const float* __restrict__ X, const float* __restrict__ W,
               const float* __restrict__ bias, float* __restrict__ Y,
               int M, int N, int K) {
    const int m0   = blockIdx.y * 64;
    const int n0   = blockIdx.x * 64;
    const int tid  = threadIdx.x;
    const int wave = tid >> 5;
    const int lane = tid & 31;

    __shared__ float Asf[2][64 * 32];   // [buf][row*32 + k]
    __shared__ float Bsf[2][32 * 64];   // [buf][k*64 + col]

    const int nIter = K >> 5;
    if (wave == 0) {   // stage tile 0 via TDM
        tdm_load_2d(X + (size_t)m0 * K, lds_off_of(&Asf[0][0]), 32, 64, (unsigned)K);
        tdm_load_2d(W + n0,             lds_off_of(&Bsf[0][0]), 64, 32, (unsigned)N);
    }

    v8f acc[4];
#pragma unroll
    for (int j = 0; j < 4; ++j) acc[j] = vzero8();

    for (int it = 0; it < nIter; ++it) {
        if (wave == 0) __builtin_amdgcn_s_wait_tensorcnt(0);  // tile(it) landed
        __syncthreads();
        if (wave == 0 && it + 1 < nIter) {                    // prefetch tile(it+1)
            const int k0n = (it + 1) * 32;
            const int nb  = (it + 1) & 1;
            tdm_load_2d(X + (size_t)m0 * K + k0n, lds_off_of(&Asf[nb][0]), 32, 64, (unsigned)K);
            tdm_load_2d(W + (size_t)k0n * N + n0, lds_off_of(&Bsf[nb][0]), 64, 32, (unsigned)N);
        }
        const float* As = Asf[it & 1];
        const float* Bs = Bsf[it & 1];

        v16bf a = frag_rowK(As, 32, wave * 16, 0, lane);
#pragma unroll
        for (int j = 0; j < 4; ++j) {
            v16bf b = frag_colK(Bs, 64, 0, j * 16, lane);
            acc[j] = __builtin_amdgcn_wmma_f32_16x16x32_bf16(
                false, a, false, b, (short)0, acc[j], false, false);
        }
        __syncthreads();
    }

    const int hi = (lane >= 16) ? 8 : 0;
#pragma unroll
    for (int j = 0; j < 4; ++j) {
        int n = n0 + j * 16 + (lane & 15);
        float bv = bias ? bias[n] : 0.0f;
#pragma unroll
        for (int r = 0; r < 8; ++r) {
            int m = m0 + wave * 16 + r + hi;
            float v = acc[j][r] + bv;
            if (EPI == 1) v = 0.5f * v * (1.0f + erff(v * 0.70710678118f));
            Y[(size_t)m * N + n] = v;
        }
    }
}

// ---------------------------------------------------------------------------
// beta = sigmoid(hs @ Wb) : one block per row, wave w computes head w.
// ---------------------------------------------------------------------------
__global__ void __launch_bounds__(256)
beta_kernel(const float* __restrict__ hs, const float* __restrict__ Wb,
            float* __restrict__ beta) {
    int row  = blockIdx.x;
    int wave = threadIdx.x >> 5, lane = threadIdx.x & 31;
    const float* x = hs + (size_t)row * DD;
    float s = 0.0f;
    for (int i = lane; i < DD; i += 32) s += x[i] * Wb[(size_t)i * HH + wave];
#pragma unroll
    for (int off = 16; off > 0; off >>= 1) s += __shfl_xor(s, off, 32);
    if (lane == 0) beta[(size_t)row * HH + wave] = 1.0f / (1.0f + __expf(-s));
}

// ---------------------------------------------------------------------------
// Depthwise causal conv over time, optional SiLU. Covers conv_q/k/v (taps=4,
// silu=1) and FIR short/long (taps=3/63, silu=0). Channel-major weights [D,taps].
// ---------------------------------------------------------------------------
__global__ void __launch_bounds__(256)
dwconv_kernel(const float* __restrict__ x, const float* __restrict__ w,
              float* __restrict__ y, int taps, int do_silu) {
    long long idx = (long long)blockIdx.x * 256 + threadIdx.x;  // over B*L*D
    if (idx >= (long long)BL * DD) return;
    int c = (int)(idx % DD);
    int t = (int)((idx / DD) % LL);
    float s = 0.0f;
    for (int j = 0; j < taps; ++j) {
        int dt = j - (taps - 1);
        if (t + dt >= 0) s += x[idx + (long long)dt * DD] * w[c * taps + j];
    }
    if (do_silu) s = s / (1.0f + __expf(-s));
    y[idx] = s;
}

// ---------------------------------------------------------------------------
// Delta rule: one workgroup per (b,h), 256 threads (8 waves).
// State S[128x128] f32 lives in dynamic LDS for the whole 128-chunk scan.
// Chunk q/k/v tiles are staged by the TDM; all 32x*x128 matmuls use bf16 WMMA.
// ---------------------------------------------------------------------------
__global__ void __launch_bounds__(256)
delta_rule_kernel(const float* __restrict__ q, const float* __restrict__ k,
                  const float* __restrict__ v, const float* __restrict__ beta,
                  const float* __restrict__ ret_param, float* __restrict__ dout) {
    extern __shared__ float sm[];
    float* S    = sm;                 // 128*128
    float* qf   = S    + 128 * 128;   // 32*128
    float* kf   = qf   + 32 * 128;
    float* kbf  = kf   + 32 * 128;
    float* vf   = kbf  + 32 * 128;
    float* ubuf = vf   + 32 * 128;    // u, then u_adj in place
    float* wbuf = ubuf + 32 * 128;
    float* Abuf = wbuf + 32 * 128;    // 32*32 (A, then attn)
    float* Tbuf = Abuf + 32 * 32;     // 32*32
    float* bc   = Tbuf + 32 * 32;     // 32

    const int bh   = blockIdx.x;
    const int b    = bh / HH, h = bh % HH;
    const int tid  = threadIdx.x;
    const int wave = tid >> 5, lane = tid & 31;
    const int hi   = (lane >= 16) ? 8 : 0;
    const int l15  = lane & 15;
    const float lam = 0.6f + 0.4f / (1.0f + __expf(-ret_param[h]));

    for (int i = tid; i < 128 * 128; i += 256) S[i] = 0.0f;
    __syncthreads();

    for (int n = 0; n < NCHUNK; ++n) {
        const size_t rowbase = (size_t)b * LL + (size_t)n * 32;
        // ---- stage chunk q/k/v [32 x 128] via TDM, beta [32] via VMEM ----
        if (wave == 0) {
            const size_t g = rowbase * DD + (size_t)h * DKV;
            tdm_load_2d(q + g, lds_off_of(qf), 128, 32, DD);
            tdm_load_2d(k + g, lds_off_of(kf), 128, 32, DD);
            tdm_load_2d(v + g, lds_off_of(vf), 128, 32, DD);
            __builtin_amdgcn_s_wait_tensorcnt(0);
        }
        if (tid < 32) bc[tid] = beta[(rowbase + tid) * HH + h];
        __syncthreads();
        // ---- l2-normalize rows of q and k ----
        if (tid < 64) {
            int c = tid & 31;
            float* buf = (tid < 32) ? qf : kf;
            float s = 0.0f;
            for (int dd = 0; dd < 128; ++dd) { float x = buf[c * 128 + dd]; s += x * x; }
            float r = rsqrtf(s + 1e-6f);
            for (int dd = 0; dd < 128; ++dd) buf[c * 128 + dd] *= r;
        }
        __syncthreads();
        // ---- kb = k*beta ; v *= beta ----
        for (int i = tid; i < 32 * 128; i += 256) {
            int c = i >> 7;
            kbf[i] = kf[i] * bc[c];
            vf[i] *= bc[c];
        }
        __syncthreads();
        // ---- A = tril(kb @ k^T, -1)  (4 WMMA tiles, waves 0..3) ----
        if (wave < 4) {
            int mi = (wave >> 1) * 16, nj = (wave & 1) * 16;
            v8f acc = vzero8();
#pragma unroll
            for (int kk = 0; kk < 128; kk += 32) {
                v16bf a = frag_rowK(kbf, 128, mi, kk, lane);
                v16bf bb = frag_rowK(kf, 128, nj, kk, lane);   // B = k^T
                acc = __builtin_amdgcn_wmma_f32_16x16x32_bf16(
                    false, a, false, bb, (short)0, acc, false, false);
            }
#pragma unroll
            for (int r = 0; r < 8; ++r) {
                int m = mi + r + hi, e = nj + l15;
                Abuf[m * 32 + e] = (e < m) ? acc[r] : 0.0f;
            }
        }
        __syncthreads();
        // ---- T = (I + A)^-1 via forward substitution, column per thread ----
        if (tid < 32) {
            int j = tid;
            for (int i = 0; i < j; ++i) Tbuf[i * 32 + j] = 0.0f;
            Tbuf[j * 32 + j] = 1.0f;
            for (int i = j + 1; i < 32; ++i) {
                float s = 0.0f;
                for (int m = j; m < i; ++m) s += Abuf[i * 32 + m] * Tbuf[m * 32 + j];
                Tbuf[i * 32 + j] = -s;
            }
        }
        __syncthreads();
        // ---- u = T@v ; w = T@kb  (16 tiles each; 2 per wave) ----
#pragma unroll
        for (int t = 0; t < 2; ++t) {
            int tt = wave * 2 + t;
            int mi = (tt >> 3) * 16, nj = (tt & 7) * 16;
            v16bf ta = frag_rowK(Tbuf, 32, mi, 0, lane);
            v16bf bv = frag_colK(vf, 128, 0, nj, lane);
            v16bf bk = frag_colK(kbf, 128, 0, nj, lane);
            v8f au = __builtin_amdgcn_wmma_f32_16x16x32_bf16(
                false, ta, false, bv, (short)0, vzero8(), false, false);
            v8f aw = __builtin_amdgcn_wmma_f32_16x16x32_bf16(
                false, ta, false, bk, (short)0, vzero8(), false, false);
#pragma unroll
            for (int r = 0; r < 8; ++r) {
                int m = mi + r + hi, c = nj + l15;
                ubuf[m * 128 + c] = au[r];
                wbuf[m * 128 + c] = aw[r];
            }
        }
        __syncthreads();
        // ---- attn = tril(q @ k^T)  (incl. diagonal), reuse Abuf ----
        if (wave < 4) {
            int mi = (wave >> 1) * 16, nj = (wave & 1) * 16;
            v8f acc = vzero8();
#pragma unroll
            for (int kk = 0; kk < 128; kk += 32) {
                v16bf a = frag_rowK(qf, 128, mi, kk, lane);
                v16bf bb = frag_rowK(kf, 128, nj, kk, lane);
                acc = __builtin_amdgcn_wmma_f32_16x16x32_bf16(
                    false, a, false, bb, (short)0, acc, false, false);
            }
#pragma unroll
            for (int r = 0; r < 8; ++r) {
                int m = mi + r + hi, e = nj + l15;
                Abuf[m * 32 + e] = (e <= m) ? acc[r] : 0.0f;
            }
        }
        __syncthreads();
        // ---- u_adj = u - w @ S   (in place into ubuf) ----
#pragma unroll
        for (int t = 0; t < 2; ++t) {
            int tt = wave * 2 + t;
            int mi = (tt >> 3) * 16, nj = (tt & 7) * 16;
            v8f acc = vzero8();
#pragma unroll
            for (int kk = 0; kk < 128; kk += 32) {
                v16bf a = frag_rowK(wbuf, 128, mi, kk, lane);
                v16bf bb = frag_colK(S, 128, kk, nj, lane);
                acc = __builtin_amdgcn_wmma_f32_16x16x32_bf16(
                    false, a, false, bb, (short)0, acc, false, false);
            }
#pragma unroll
            for (int r = 0; r < 8; ++r) {
                int m = mi + r + hi, c = nj + l15;
                ubuf[m * 128 + c] -= acc[r];
            }
        }
        __syncthreads();
        // ---- o = q @ S + attn @ u_adj -> global ----
#pragma unroll
        for (int t = 0; t < 2; ++t) {
            int tt = wave * 2 + t;
            int mi = (tt >> 3) * 16, nj = (tt & 7) * 16;
            v8f acc = vzero8();
#pragma unroll
            for (int kk = 0; kk < 128; kk += 32) {
                v16bf a = frag_rowK(qf, 128, mi, kk, lane);
                v16bf bb = frag_colK(S, 128, kk, nj, lane);
                acc = __builtin_amdgcn_wmma_f32_16x16x32_bf16(
                    false, a, false, bb, (short)0, acc, false, false);
            }
            {
                v16bf a = frag_rowK(Abuf, 32, mi, 0, lane);
                v16bf bb = frag_colK(ubuf, 128, 0, nj, lane);
                acc = __builtin_amdgcn_wmma_f32_16x16x32_bf16(
                    false, a, false, bb, (short)0, acc, false, false);
            }
#pragma unroll
            for (int r = 0; r < 8; ++r) {
                int m = mi + r + hi, c = nj + l15;
                dout[(rowbase + m) * DD + h * DKV + c] = acc[r];
            }
        }
        __syncthreads();
        // ---- S = lam*S + k^T @ u_adj  (64 tiles; 8 per wave) ----
#pragma unroll
        for (int t = 0; t < 8; ++t) {
            int tt = wave * 8 + t;
            int mi = (tt >> 3) * 16, nj = (tt & 7) * 16;
            v8f acc;
#pragma unroll
            for (int r = 0; r < 8; ++r)
                acc[r] = lam * S[(mi + r + hi) * 128 + nj + l15];
            v16bf a = frag_colK(kf, 128, 0, mi, lane);      // A = k^T
            v16bf bb = frag_colK(ubuf, 128, 0, nj, lane);   // B = u_adj
            acc = __builtin_amdgcn_wmma_f32_16x16x32_bf16(
                false, a, false, bb, (short)0, acc, false, false);
#pragma unroll
            for (int r = 0; r < 8; ++r)
                S[(mi + r + hi) * 128 + nj + l15] = acc[r];
        }
        __syncthreads();
    }
}

// ---------------------------------------------------------------------------
// feat = concat(hs, mean_dv(f_short), mean_dv(f_long), mean_dv(delta), mean_dv(v))
// ---------------------------------------------------------------------------
__global__ void __launch_bounds__(256)
feat_kernel(const float* __restrict__ hs, const float* __restrict__ fs,
            const float* __restrict__ fl, const float* __restrict__ dl,
            const float* __restrict__ vd, float* __restrict__ feat) {
    const int FD = DD + 4 * HH;   // 1056
    int row = blockIdx.x, tid = threadIdx.x;
    for (int i = tid; i < DD; i += 256)
        feat[(size_t)row * FD + i] = hs[(size_t)row * DD + i];
    if (tid < 32) {
        int which = tid >> 3, h = tid & 7;
        const float* src = (which == 0) ? fs : (which == 1) ? fl : (which == 2) ? dl : vd;
        const float* p = src + (size_t)row * DD + h * DKV;
        float s = 0.0f;
        for (int dd = 0; dd < DKV; ++dd) s += p[dd];
        feat[(size_t)row * FD + DD + which * HH + h] = s * (1.0f / 128.0f);
    }
}

// ---------------------------------------------------------------------------
// p = eps + (1-eps)*sigmoid((hdn @ gate_w2 + b2)/exp(log_temp)) -> pbuf[BL,32]
// ---------------------------------------------------------------------------
__global__ void __launch_bounds__(256)
gate2_kernel(const float* __restrict__ hdn, const float* __restrict__ w2,
             const float* __restrict__ b2, const float* __restrict__ log_temp,
             float* __restrict__ pbuf) {
    int row  = blockIdx.x;
    int wave = threadIdx.x >> 5, lane = threadIdx.x & 31;
    const float* x = hdn + (size_t)row * 2048;
    for (int o = wave; o < 32; o += 8) {
        float s = 0.0f;
        for (int i = lane; i < 2048; i += 32) s += x[i] * w2[(size_t)i * 32 + o];
#pragma unroll
        for (int off = 16; off > 0; off >>= 1) s += __shfl_xor(s, off, 32);
        if (lane == 0) {
            s = (s + b2[o]) * __expf(-log_temp[o >> 2]);
            pbuf[(size_t)row * 32 + o] = 0.02f + 0.98f / (1.0f + __expf(-s));
        }
    }
}

// ---------------------------------------------------------------------------
// o = p0*fs + p1*fl + p2*delta + p3*v ; RMS-norm over dv ; * o_norm_w
// one block per (row, h), 128 threads (one per dv).
// ---------------------------------------------------------------------------
__global__ void __launch_bounds__(128)
mix_norm_kernel(const float* __restrict__ fs, const float* __restrict__ fl,
                const float* __restrict__ dl, const float* __restrict__ vd,
                const float* __restrict__ pbuf, const float* __restrict__ onw,
                float* __restrict__ omix) {
    int bh = blockIdx.x;
    int row = bh >> 3, h = bh & 7;
    int dv = threadIdx.x;
    size_t base = (size_t)row * DD + h * DKV + dv;
    const float* p = pbuf + (size_t)row * 32 + h * 4;
    float o = p[0] * fs[base] + p[1] * fl[base] + p[2] * dl[base] + p[3] * vd[base];
    __shared__ float red[128];
    red[dv] = o * o;
    __syncthreads();
    for (int s = 64; s > 0; s >>= 1) {
        if (dv < s) red[dv] += red[dv + s];
        __syncthreads();
    }
    float ms = red[0] * (1.0f / 128.0f);
    omix[base] = o * rsqrtf(ms + 1e-5f) * onw[dv];
}

// ---------------------------------------------------------------------------
extern "C" void kernel_launch(void* const* d_in, const int* in_sizes, int n_in,
                              void* d_out, int out_size, void* d_ws, size_t ws_size,
                              hipStream_t stream) {
    (void)in_sizes; (void)n_in; (void)out_size; (void)ws_size;
    const float* hs        = (const float*)d_in[0];
    const float* Wq        = (const float*)d_in[1];
    const float* Wk        = (const float*)d_in[2];
    const float* Wv        = (const float*)d_in[3];
    const float* Wb        = (const float*)d_in[4];
    const float* conv_q_w  = (const float*)d_in[5];
    const float* conv_k_w  = (const float*)d_in[6];
    const float* conv_v_w  = (const float*)d_in[7];
    const float* ret_param = (const float*)d_in[8];
    const float* fir_s_w   = (const float*)d_in[9];
    const float* fir_l_w   = (const float*)d_in[10];
    const float* gate_w1   = (const float*)d_in[11];
    const float* gate_b1   = (const float*)d_in[12];
    const float* gate_w2   = (const float*)d_in[13];
    const float* gate_b2   = (const float*)d_in[14];
    const float* log_temp  = (const float*)d_in[15];
    const float* o_norm_w  = (const float*)d_in[16];
    const float* Wo        = (const float*)d_in[17];
    float* out = (float*)d_out;

    // ---- workspace carving (floats) ----
    const size_t BLD = (size_t)BL * DD;
    float* ws = (float*)d_ws;
    size_t off = 0;
    float* buf_a = ws + off; off += BLD;              // q_lin -> f_short
    float* buf_b = ws + off; off += BLD;              // k_lin -> f_long
    float* buf_c = ws + off; off += BLD;              // v_lin -> o_mixed
    float* qb    = ws + off; off += BLD;              // q post conv
    float* kb    = ws + off; off += BLD;              // k post conv
    float* vb    = ws + off; off += BLD;              // v post conv
    float* dlt   = ws + off; off += BLD;              // delta_out
    float* feat  = ws + off; off += (size_t)BL * (DD + 4 * HH);
    float* hdn   = ws + off; off += (size_t)BL * 2048;
    float* betab = ws + off; off += (size_t)BL * HH;
    float* pbuf  = ws + off; off += (size_t)BL * 32;

    const dim3 gemm_blk(128);
    const dim3 g_proj(DD / 64, BL / 64);        // 16 x 256
    const dim3 g_gate1(2048 / 64, BL / 64);     // 32 x 256
    const int  conv_blocks = (int)(BLD / 256);

    // 1) q/k/v linear projections (TDM-staged bf16 WMMA GEMM)
    gemm_bf16_wmma<0><<<g_proj, gemm_blk, 0, stream>>>(hs, Wq, nullptr, buf_a, BL, DD, DD);
    gemm_bf16_wmma<0><<<g_proj, gemm_blk, 0, stream>>>(hs, Wk, nullptr, buf_b, BL, DD, DD);
    gemm_bf16_wmma<0><<<g_proj, gemm_blk, 0, stream>>>(hs, Wv, nullptr, buf_c, BL, DD, DD);
    // 2) beta
    beta_kernel<<<BL, 256, 0, stream>>>(hs, Wb, betab);
    // 3) short convs + silu
    dwconv_kernel<<<conv_blocks, 256, 0, stream>>>(buf_a, conv_q_w, qb, 4, 1);
    dwconv_kernel<<<conv_blocks, 256, 0, stream>>>(buf_b, conv_k_w, kb, 4, 1);
    dwconv_kernel<<<conv_blocks, 256, 0, stream>>>(buf_c, conv_v_w, vb, 4, 1);
    // 4) delta rule (state in LDS, TDM chunk staging, WMMA chunk matmuls)
    const size_t smem = (size_t)(128 * 128 + 6 * 32 * 128 + 2 * 32 * 32 + 32) * sizeof(float);
    delta_rule_kernel<<<BB * HH, 256, smem, stream>>>(qb, kb, vb, betab, ret_param, dlt);
    // 5) FIR filters on v (reuse buf_a / buf_b)
    dwconv_kernel<<<conv_blocks, 256, 0, stream>>>(vb, fir_s_w, buf_a, 3, 0);
    dwconv_kernel<<<conv_blocks, 256, 0, stream>>>(vb, fir_l_w, buf_b, 63, 0);
    // 6) features + gate MLP
    feat_kernel<<<BL, 256, 0, stream>>>(hs, buf_a, buf_b, dlt, vb, feat);
    gemm_bf16_wmma<1><<<g_gate1, gemm_blk, 0, stream>>>(feat, gate_w1, gate_b1, hdn,
                                                        BL, 2048, DD + 4 * HH);
    gate2_kernel<<<BL, 256, 0, stream>>>(hdn, gate_w2, gate_b2, log_temp, pbuf);
    // 7) mix + RMS norm (into buf_c), final output projection
    mix_norm_kernel<<<BL * HH, 128, 0, stream>>>(buf_a, buf_b, dlt, vb, pbuf, o_norm_w, buf_c);
    gemm_bf16_wmma<0><<<g_proj, gemm_blk, 0, stream>>>(buf_c, Wo, nullptr, out, BL, DD, DD);
}
